// GemAttention_54202487275853
// MI455X (gfx1250) — matrix operand order repaired and checked by reading, not compiled, not run on hardware
//
#include <hip/hip_runtime.h>
#include <hip/hip_bf16.h>

// GEM (power-mean) attention for MI455X / gfx1250, wave32, WMMA bf16 + TDM.
// Pipeline: prep -> gemm_qkv -> minmax -> postqkv -> attn (fused) -> gemm_out
// Workspace use: ~6.3 MB of d_ws.

#define D_DIM   256
#define H_DIM   8
#define A_DIM   32
#define S_DIM   512
#define B_DIM   2
#define NROW    1024      // B*S
#define NQKV    768       // q(256) | k(256) | v(256) columns of fused QKV GEMM
#define B_CONSTF 0.9f
#define P_MAXF   50000.0f
#define P_MINF   0.0001f
#define TANH_RATEF 0.005f
#define EPS_CF   1e-10f
#define SCALE_CF 0.0625f  // 1/sqrt(256)

typedef __bf16 bf16_t;
typedef __attribute__((ext_vector_type(16))) __bf16 v16bf;
typedef __attribute__((ext_vector_type(8)))  float  v8f;
typedef __attribute__((ext_vector_type(4)))  unsigned int uint32x4;
typedef __attribute__((ext_vector_type(8)))  int int32x8;
typedef __attribute__((ext_vector_type(4)))  int int32x4;

#if defined(__has_builtin)
#if __has_builtin(__builtin_amdgcn_tensor_load_to_lds)
#define HAVE_TDM 1
#else
#define HAVE_TDM 0
#endif
#if __has_builtin(__builtin_amdgcn_s_wait_tensorcnt)
#define HAVE_TCNT 1
#else
#define HAVE_TCNT 0
#endif
#else
#define HAVE_TDM 0
#define HAVE_TCNT 0
#endif

union V16U { v16bf v; uint4 q[2]; };

// B-fragment: 16 contiguous bf16 (one row chunk of the "N x K" operand).
__device__ __forceinline__ v16bf load_b_row16(const bf16_t* p) {
  V16U u;
  u.q[0] = *reinterpret_cast<const uint4*>(p);
  u.q[1] = *reinterpret_cast<const uint4*>(p + 8);
  return u.v;
}
// A-fragment: two contiguous 8-element bf16 chunks (K and K+8).
__device__ __forceinline__ v16bf load_a_2chunks(const bf16_t* p0, const bf16_t* p1) {
  V16U u;
  u.q[0] = *reinterpret_cast<const uint4*>(p0);
  u.q[1] = *reinterpret_cast<const uint4*>(p1);
  return u.v;
}

__device__ __forceinline__ v8f wmma_bf16(v16bf a, v16bf b, v8f c) {
  // D = A(16x32 bf16) x B(32x16 bf16) + C(16x16 f32)
  return __builtin_amdgcn_wmma_f32_16x16x32_bf16(
      /*neg_a=*/false, a, /*neg_b=*/false, b,
      /*c_mod=*/(short)0, c, /*reuse_a=*/false, /*reuse_b=*/false);
}

#if HAVE_TDM
// TDM: DMA a 2D bf16 tile (32 rows x 512 cols, row stride 512 elems) into LDS.
// D# group layout per CDNA5 ISA ch.8 (CDNA5 ISA).
// 6-arg builtin form (this toolchain): (g0, g1, g2, g3, g4, cpol).
__device__ __forceinline__ void tdm_load_vp(const bf16_t* gsrc, void* lds_dst) {
  unsigned long long ga = (unsigned long long)(uintptr_t)gsrc;
  unsigned int ldsoff = (unsigned int)(uintptr_t)lds_dst;  // low 32b = LDS offset
  uint32x4 g0;
  g0[0] = 1u;                                         // count=1, user mode
  g0[1] = ldsoff;                                     // lds_addr
  g0[2] = (unsigned int)(ga & 0xffffffffull);         // global_addr[31:0]
  g0[3] = (unsigned int)((ga >> 32) & 0x1ffffffull)   // global_addr[56:32]
          | (2u << 30);                               // type=2 ("image")
  int32x8 g1;
  g1[0] = (int)(1u << 16);    // data_size=1 (2 bytes/elem)
  g1[1] = (int)(512u << 16);  // tensor_dim0[15:0] = 512
  g1[2] = (int)(32u << 16);   // tensor_dim1[15:0] = 32
  g1[3] = (int)(512u << 16);  // tile_dim0 = 512
  g1[4] = 32;                 // tile_dim1 = 32 (tile_dim2 = 0 -> 2D)
  g1[5] = 512;                // tensor_dim0_stride[31:0] = 512
  g1[6] = 0;
  g1[7] = 0;
  int32x4 z4 = {0, 0, 0, 0};          // groups 2/3 unused for 2D tensor
  int32x8 z8 = {0, 0, 0, 0, 0, 0, 0, 0};
  __builtin_amdgcn_tensor_load_to_lds(g0, g1, z4, z4, z8, 0);
}
#endif

// ---------------------------------------------------------------------------
// prep: bf16 casts of context / [W_Q;W_KV] / W_out, and p = 5e4*tanh(5e-3*pp)+1
// ---------------------------------------------------------------------------
extern "C" __global__ void prep_kernel(const float* __restrict__ ctx,
                                       const float* __restrict__ WQ,
                                       const float* __restrict__ WKV,
                                       const float* __restrict__ Wout,
                                       const float* __restrict__ pp,
                                       bf16_t* __restrict__ xbf,
                                       bf16_t* __restrict__ wcat,
                                       bf16_t* __restrict__ woutbf,
                                       float* __restrict__ pvec) {
  int gid = blockIdx.x * blockDim.x + threadIdx.x;
  int stride = gridDim.x * blockDim.x;
  for (int i = gid; i < NROW * D_DIM; i += stride) xbf[i] = (bf16_t)ctx[i];
  for (int i = gid; i < D_DIM * D_DIM; i += stride) wcat[i] = (bf16_t)WQ[i];
  for (int i = gid; i < 2 * D_DIM * D_DIM; i += stride)
    wcat[D_DIM * D_DIM + i] = (bf16_t)WKV[i];
  for (int i = gid; i < D_DIM * D_DIM; i += stride) woutbf[i] = (bf16_t)Wout[i];
  for (int i = gid; i < D_DIM; i += stride) {
    float t = P_MAXF * tanhf(TANH_RATEF * pp[i]) + 1.0f;
    if (t == 0.0f) t += P_MINF;
    pvec[i] = t;
  }
}

// ---------------------------------------------------------------------------
// gemm_qkv: out1(1024x768 f32) = Xbf(1024x256) @ Wcat(768x256)^T  via WMMA
// One wave per 16x64 strip: one A-fragment feeds 4 WMMAs per K-step.
// ---------------------------------------------------------------------------
extern "C" __global__ void gemm_qkv_kernel(const bf16_t* __restrict__ xbf,
                                           const bf16_t* __restrict__ wcat,
                                           float* __restrict__ out1) {
  const int l  = threadIdx.x;
  const int lm = l & 15;
  const int hi = (l < 16) ? 0 : 1;
  const int m0 = blockIdx.x * 16;
  const int n0 = blockIdx.y * 64;
  const bf16_t* arow = xbf  + (m0 + lm) * D_DIM;
  const bf16_t* brow = wcat + (n0 + lm) * D_DIM;
  __builtin_prefetch(arow, 0, 1);
  __builtin_prefetch(brow, 0, 1);
  v8f acc[4] = {{}, {}, {}, {}};
#pragma unroll
  for (int kk = 0; kk < D_DIM; kk += 32) {
    v16bf a = load_a_2chunks(arow + kk + hi * 16, arow + kk + hi * 16 + 8);
#pragma unroll
    for (int t = 0; t < 4; ++t) {
      v16bf b = load_b_row16(brow + t * 16 * D_DIM + kk + hi * 16);
      acc[t] = wmma_bf16(a, b, acc[t]);
    }
  }
#pragma unroll
  for (int t = 0; t < 4; ++t)
#pragma unroll
    for (int r = 0; r < 8; ++r)
      out1[(m0 + r + hi * 8) * NQKV + n0 + t * 16 + lm] = acc[t][r];
}

// ---------------------------------------------------------------------------
// minmax: per (batch, channel) min/max of v over S
// ---------------------------------------------------------------------------
extern "C" __global__ void minmax_kernel(const float* __restrict__ out1,
                                         float* __restrict__ vmin,
                                         float* __restrict__ vmax) {
  int b = blockIdx.x;
  int d = threadIdx.x;
  float mn = 3.4e38f, mx = -3.4e38f;
  for (int s = 0; s < S_DIM; ++s) {
    float v = out1[(b * S_DIM + s) * NQKV + 512 + d];
    mn = fminf(mn, v);
    mx = fmaxf(mx, v);
  }
  vmin[b * D_DIM + d] = mn;
  vmax[b * D_DIM + d] = mx;
}

// ---------------------------------------------------------------------------
// postqkv: qs = q/16 (bf16), k (bf16), VpT[b][d][s] = v_norm^p (bf16, transposed)
// ---------------------------------------------------------------------------
extern "C" __global__ void postqkv_kernel(const float* __restrict__ out1,
                                          const float* __restrict__ pvec,
                                          const float* __restrict__ vmin,
                                          const float* __restrict__ vmax,
                                          bf16_t* __restrict__ qsbf,
                                          bf16_t* __restrict__ kbf,
                                          bf16_t* __restrict__ vptbf) {
  int gid = blockIdx.x * blockDim.x + threadIdx.x;
  int stride = gridDim.x * blockDim.x;
  for (int i = gid; i < NROW * D_DIM; i += stride) {
    int row = i >> 8;          // /256
    int d   = i & 255;
    int b   = row >> 9;        // /512
    int s   = row & 511;
    const float* o = out1 + row * NQKV;
    qsbf[i] = (bf16_t)(o[d] * SCALE_CF);
    kbf[i]  = (bf16_t)(o[256 + d]);
    float vr  = o[512 + d];
    float mn  = vmin[b * D_DIM + d];
    float rng = vmax[b * D_DIM + d] - mn + EPS_CF;
    float vn  = (1.0f - B_CONSTF) * (vr - mn) / rng + B_CONSTF;   // in [0.9, 1]
    float vp  = expf(pvec[d] * logf(vn));                          // v_norm^p
    vptbf[(b * D_DIM + d) * S_DIM + s] = (bf16_t)vp;
  }
}

// ---------------------------------------------------------------------------
// attn: per (b, h, 16-row q-tile), one wave.
//   TDM prefetch of Vp head slice (32x512 bf16) into LDS, overlapped with:
//   scores(16x512) = Q(16x32) @ K^T via 32 WMMAs -> LDS (f32)
//   row max/sum -> packed-bf16 probs in LDS -> s_wait_tensorcnt
//   M(16x32) = P @ Vp via 2x16 WMMAs (A + B fragments all ds_load_b128)
//   epilogue: mean = log(M/sum); e = unnormalize(exp(mean/p)) -> bf16
// ---------------------------------------------------------------------------
extern "C" __global__ void attn_kernel(const bf16_t* __restrict__ qsbf,
                                       const bf16_t* __restrict__ kbf,
                                       const bf16_t* __restrict__ vptbf,
                                       const float* __restrict__ pvec,
                                       const float* __restrict__ vmin,
                                       const float* __restrict__ vmax,
                                       bf16_t* __restrict__ ebf) {
  __shared__ __align__(16) float  sm_scores[16][S_DIM];   // 32 KB
  __shared__ __align__(16) bf16_t sm_probs[16][S_DIM];    // 16 KB
  __shared__ __align__(16) bf16_t sm_vp[32][S_DIM];       // 32 KB (TDM target)
  __shared__ float sm_sum[16];

  const int l  = threadIdx.x;
  const int lm = l & 15;
  const int hi = (l < 16) ? 0 : 1;
  const int qt = blockIdx.x, h = blockIdx.y, b = blockIdx.z;
  const int m0 = qt * 16;
  const int cbase = h * A_DIM;
  const bf16_t* vpsrc = vptbf + (b * D_DIM + cbase) * S_DIM;

#if HAVE_TDM
  // Kick off async Vp staging; overlaps with the 32 score WMMAs below.
  tdm_load_vp(vpsrc, &sm_vp[0][0]);
#else
  for (int i = l * 8; i < 32 * S_DIM; i += 32 * 8)
    *reinterpret_cast<uint4*>(&sm_vp[0][0] + i) =
        *reinterpret_cast<const uint4*>(vpsrc + i);
#endif

  // Q A-fragment (reused across all 32 score WMMAs)
  const bf16_t* qrow = qsbf + (b * S_DIM + m0 + lm) * D_DIM + cbase;
  v16bf aq = load_a_2chunks(qrow + hi * 16, qrow + hi * 16 + 8);

  // pass 1: scaled scores
  for (int kt = 0; kt < 32; ++kt) {
    int k0 = kt * 16;
    const bf16_t* krow =
        kbf + (b * S_DIM + k0 + lm) * D_DIM + cbase + hi * 16;
    v8f c = {};
    c = wmma_bf16(aq, load_b_row16(krow), c);
#pragma unroll
    for (int r = 0; r < 8; ++r)
      sm_scores[r + hi * 8][k0 + lm] = c[r];
  }
  __syncthreads();

  // softmax stats: lane (lm, hi) scans half of row lm, combine halves via shfl
  float rmax = -3.4e38f;
  for (int i = 0; i < 256; ++i)
    rmax = fmaxf(rmax, sm_scores[lm][hi * 256 + i]);
  rmax = fmaxf(rmax, __shfl_xor(rmax, 16));
  float rsum = 0.0f;
  for (int i = 0; i < 256; ++i)
    rsum += expf(sm_scores[lm][hi * 256 + i] - rmax);
  rsum += __shfl_xor(rsum, 16);
  if (l < 16) sm_sum[lm] = rsum;
  // exponentiate -> packed bf16 probs (ds_store_b32 pairs)
  for (int i = 0; i < 128; ++i) {
    int k = hi * 256 + 2 * i;
    float e0 = expf(sm_scores[lm][k] - rmax);
    float e1 = expf(sm_scores[lm][k + 1] - rmax);
    union { bf16_t bb[2]; unsigned int u; } pk;
    pk.bb[0] = (bf16_t)e0;
    pk.bb[1] = (bf16_t)e1;
    *reinterpret_cast<unsigned int*>(&sm_probs[lm][k]) = pk.u;
  }
#if HAVE_TDM && HAVE_TCNT
  __builtin_amdgcn_s_wait_tensorcnt(0);  // Vp stage complete before pass 2
#endif
  __syncthreads();

  // pass 2: M(16x32) = P(16x512) @ Vp(512x32); all fragments from LDS
  v8f acc0 = {}, acc1 = {};
  const bf16_t* prow = &sm_probs[lm][0];
  for (int kb = 0; kb < 16; ++kb) {
    int k0 = kb * 32;
    int base0 = k0 + hi * 16;
    v16bf ap = load_a_2chunks(prow + base0, prow + base0 + 8);
    const bf16_t* b0 = &sm_vp[lm][k0 + hi * 16];
    const bf16_t* b1 = &sm_vp[16 + lm][k0 + hi * 16];
    acc0 = wmma_bf16(ap, load_b_row16(b0), acc0);
    acc1 = wmma_bf16(ap, load_b_row16(b1), acc1);
  }

  // epilogue
#pragma unroll
  for (int r = 0; r < 8; ++r) {
    int m = r + hi * 8;
    int rowg = b * S_DIM + m0 + m;
    float rs = sm_sum[m];
#pragma unroll
    for (int cc = 0; cc < 2; ++cc) {
      float msum = (cc == 0) ? acc0[r] : acc1[r];
      int d = cbase + cc * 16 + lm;
      float mean = logf(msum / rs);
      float e = expf(mean / pvec[d]);
      float mn  = vmin[b * D_DIM + d];
      float rng = vmax[b * D_DIM + d] - mn + EPS_CF;
      e = (e - B_CONSTF) * rng / (1.0f - B_CONSTF) + mn;
      ebf[rowg * D_DIM + d] = (bf16_t)e;
    }
  }
}

// ---------------------------------------------------------------------------
// gemm_out: out(1024x256 f32) = ebf(1024x256) @ Wout(256x256)^T via WMMA
// One wave per 16x64 strip.
// ---------------------------------------------------------------------------
extern "C" __global__ void gemm_out_kernel(const bf16_t* __restrict__ ebf,
                                           const bf16_t* __restrict__ woutbf,
                                           float* __restrict__ out) {
  const int l  = threadIdx.x;
  const int lm = l & 15;
  const int hi = (l < 16) ? 0 : 1;
  const int m0 = blockIdx.x * 16;
  const int n0 = blockIdx.y * 64;
  const bf16_t* arow = ebf    + (m0 + lm) * D_DIM;
  const bf16_t* brow = woutbf + (n0 + lm) * D_DIM;
  __builtin_prefetch(arow, 0, 1);
  __builtin_prefetch(brow, 0, 1);
  v8f acc[4] = {{}, {}, {}, {}};
#pragma unroll
  for (int kk = 0; kk < D_DIM; kk += 32) {
    v16bf a = load_a_2chunks(arow + kk + hi * 16, arow + kk + hi * 16 + 8);
#pragma unroll
    for (int t = 0; t < 4; ++t) {
      v16bf b = load_b_row16(brow + t * 16 * D_DIM + kk + hi * 16);
      acc[t] = wmma_bf16(a, b, acc[t]);
    }
  }
#pragma unroll
  for (int t = 0; t < 4; ++t)
#pragma unroll
    for (int r = 0; r < 8; ++r)
      out[(m0 + r + hi * 8) * D_DIM + n0 + t * 16 + lm] = acc[t][r];
}

// ---------------------------------------------------------------------------
// launch
// ---------------------------------------------------------------------------
extern "C" void kernel_launch(void* const* d_in, const int* in_sizes, int n_in,
                              void* d_out, int out_size, void* d_ws, size_t ws_size,
                              hipStream_t stream) {
  const float* ctx  = (const float*)d_in[0];   // (2,512,256)
  const float* WQ   = (const float*)d_in[1];   // (256,256)
  const float* WKV  = (const float*)d_in[2];   // (512,256)
  const float* Wout = (const float*)d_in[3];   // (256,256)
  const float* pp   = (const float*)d_in[4];   // (256,)
  float* out        = (float*)d_out;           // (2,512,256)

  char* ws = (char*)d_ws;                       // needs ~6.3 MB
  bf16_t* xbf    = (bf16_t*)(ws + 0);           // 512 KB
  bf16_t* wcat   = (bf16_t*)(ws + 524288);      // 384 KB  [W_Q; W_KV]
  bf16_t* woutbf = (bf16_t*)(ws + 917504);      // 128 KB
  float*  out1   = (float*) (ws + 1048576);     // 3 MB    q|k|v f32
  float*  pvec   = (float*) (ws + 4194304);     // 1 KB
  float*  vminp  = (float*) (ws + 4195328);     // 2 KB
  float*  vmaxp  = (float*) (ws + 4197376);     // 2 KB
  bf16_t* qsbf   = (bf16_t*)(ws + 4199424);     // 512 KB  q * SCALE
  bf16_t* kbf    = (bf16_t*)(ws + 4723712);     // 512 KB
  bf16_t* vptbf  = (bf16_t*)(ws + 5248000);     // 512 KB  Vp transposed (B,D,S)
  bf16_t* ebf    = (bf16_t*)(ws + 5772288);     // 512 KB

  prep_kernel<<<dim3(256), dim3(256), 0, stream>>>(ctx, WQ, WKV, Wout, pp,
                                                   xbf, wcat, woutbf, pvec);
  gemm_qkv_kernel<<<dim3(NROW / 16, NQKV / 64), dim3(32), 0, stream>>>(xbf, wcat, out1);
  minmax_kernel<<<dim3(B_DIM), dim3(D_DIM), 0, stream>>>(out1, vminp, vmaxp);
  postqkv_kernel<<<dim3(256), dim3(256), 0, stream>>>(out1, pvec, vminp, vmaxp,
                                                      qsbf, kbf, vptbf);
  attn_kernel<<<dim3(S_DIM / 16, H_DIM, B_DIM), dim3(32), 0, stream>>>(
      qsbf, kbf, vptbf, pvec, vminp, vmaxp, ebf);
  gemm_out_kernel<<<dim3(NROW / 16, D_DIM / 64), dim3(32), 0, stream>>>(ebf, woutbf, out);
}
